// Net_LSV_8658654068755
// MI455X (gfx1250) — compile-verified, matrix operand order checked by reading, not run on hardware
//
#include <hip/hip_runtime.h>

typedef __attribute__((ext_vector_type(16))) _Float16 v16h;
typedef __attribute__((ext_vector_type(8)))  _Float16 v8h;
typedef __attribute__((ext_vector_type(8)))  float    v8f;
typedef __attribute__((ext_vector_type(4)))  unsigned int u32x4;
typedef __attribute__((ext_vector_type(8)))  int      i32x8;
typedef __attribute__((ext_vector_type(4)))  int      i32x4;

#define MC_N     32768
#define NSTEPS   64
#define RATE_C   0.05f
#define TPB      256
#define NWAVES   8

// d_out layout (floats), concatenated in reference return order
#define OFF_PATH 0
#define OFF_VAR  2129920
#define OFF_PMAT 4259840
#define OFF_VPM  4259866
#define OFF_EXO  4259892
#define OFF_MEAN 4292660
#define OFF_VARP 4292661
#define OFF_ERR  4292662

#if __has_builtin(__builtin_amdgcn_tensor_load_to_lds)
#define HAVE_TDM 1
#else
#define HAVE_TDM 0
#endif

struct KArgs {
  const float *S0, *z, *z2, *tg, *v0, *rho;
  // per net (jax pytree key-sorted order: Wh, Wi, Wo, bh, bi, bo)
  const float *dWh,*dWi,*dWo,*dbh,*dbi,*dbo;   // p_diff   (din=3, w=64)
  const float *vWh,*vWi,*vWo,*vbh,*vbi,*vbo;   // p_driftV (din=1, w=64)
  const float *wWh,*wWi,*wWo,*wbh,*wbi,*wbo;   // p_diffV  (din=1, w=64)
  const float *cWh,*cWi,*cWo,*cbh,*cbi,*cbo;   // p_cvv    (din=2, w=30, dout=26)
  const float *eWh,*eWi,*eWo,*ebh,*ebi,*ebo;   // p_cve    (din=67, w=20, dout=1)
  float *out; float *ws;
};

struct __align__(32) Lds {
  __align__(32) _Float16 wh[3][3][4096];     // swizzled f16 hidden weights (per net, per layer)
  __align__(32) _Float16 scr[NWAVES][32*64]; // per-wave transpose scratch (32 rows x 64 cols)
  __align__(32) float stage[3*4096];         // TDM staging: one net's 3 hidden layers (fp32)
  float tg[66];
  float wi[3][192];       // input-layer weights (net0 uses 192, others 64)
  float bi[3][64];
  float bh[3][3][64];
  float wo[3][64];
  float bo[3];
  float cvv_wi[60], cvv_bi[30], cvv_wh[2][900], cvv_bh[2][30], cvv_wo[780], cvv_bo[26];
  float cve_wi[1340], cve_bi[20], cve_wh[2][400], cve_bh[2][20], cve_wo[20], cve_bo[1];
  float svS[NWAVES][32], svV[NWAVES][32];
  float red[64];
  float path[TPB][65];
};

__device__ __forceinline__ float softplusf(float x) {
  return (x > 20.f) ? x : log1pf(expf(x));
}

__device__ __forceinline__ void cpyf(float* d, const float* s, int n, int tid) {
  for (int t = tid; t < n; t += TPB) d[t] = s[t];
}

// ---------------- WMMA 64-wide MLP hidden stack + output dot ----------------
template <int NET>
__device__ __forceinline__ float mlp64(Lds& L, int wv, int lane, v16h a[2][2]) {
  const int lo16 = lane & 15, grp = lane >> 4;
  _Float16* scr = L.scr[wv];
#pragma unroll
  for (int layer = 0; layer < 3; ++layer) {
    v8f c[2][4];
#pragma unroll
    for (int m = 0; m < 2; ++m)
#pragma unroll
      for (int nt = 0; nt < 4; ++nt) {
        v8f zz = {0.f,0.f,0.f,0.f,0.f,0.f,0.f,0.f};
        c[m][nt] = zz;
      }
#pragma unroll
    for (int ks = 0; ks < 2; ++ks)
#pragma unroll
      for (int nt = 0; nt < 4; ++nt) {
        const v16h b = *(const v16h*)&L.wh[NET][layer][(((ks << 2) | nt) * 32 + lane) * 16];
        c[0][nt] = __builtin_amdgcn_wmma_f32_16x16x32_f16(
            false, a[0][ks], false, b, (short)0, c[0][nt], false, false);
        c[1][nt] = __builtin_amdgcn_wmma_f32_16x16x32_f16(
            false, a[1][ks], false, b, (short)0, c[1][nt], false, false);
      }
    // bias + relu, scatter C-layout -> row-major scratch (f16)
#pragma unroll
    for (int m = 0; m < 2; ++m)
#pragma unroll
      for (int nt = 0; nt < 4; ++nt) {
        float bb = L.bh[NET][layer][nt * 16 + lo16];
#pragma unroll
        for (int j = 0; j < 8; ++j) {
          int r = m * 16 + grp * 8 + j;
          scr[r * 64 + nt * 16 + lo16] = (_Float16)fmaxf(c[m][nt][j] + bb, 0.f);
        }
      }
    __builtin_amdgcn_wave_barrier();
    if (layer < 2) {
      // gather A-layout fragments for the next layer
#pragma unroll
      for (int m = 0; m < 2; ++m)
#pragma unroll
        for (int ks = 0; ks < 2; ++ks) {
          const _Float16* p = scr + (m * 16 + lo16) * 64 + ks * 32 + grp * 8;
          union { v16h v; v8h h[2]; } u;
          u.h[0] = *(const v8h*)p;
          u.h[1] = *(const v8h*)(p + 16);
          a[m][ks] = u.v;
        }
      __builtin_amdgcn_wave_barrier();
    }
  }
  // output layer 64 -> 1 : dot over own row
  float s = L.bo[NET];
#pragma unroll
  for (int k8 = 0; k8 < 8; ++k8) {
    v8h hv = *(const v8h*)&scr[lane * 64 + k8 * 8];
#pragma unroll
    for (int j = 0; j < 8; ++j) s += (float)hv[j] * L.wo[NET][k8 * 8 + j];
  }
  return s;
}

// input layer builders -> A fragments (16-bit A 16x32 layout)
__device__ __forceinline__ void build_frags_diff(const Lds& L, int wv, int lo16, int grp,
                                                 float tp, v16h a[2][2]) {
#pragma unroll
  for (int m = 0; m < 2; ++m) {
    int r = m * 16 + lo16;
    float Sr = L.svS[wv][r], Vr = L.svV[wv][r];
#pragma unroll
    for (int ks = 0; ks < 2; ++ks) {
      v16h f;
#pragma unroll
      for (int e = 0; e < 16; ++e) {
        int n = ks * 32 + (e >> 3) * 16 + grp * 8 + (e & 7);
        float x = tp * L.wi[0][n] + Sr * L.wi[0][64 + n] + Vr * L.wi[0][128 + n] + L.bi[0][n];
        f[e] = (_Float16)fmaxf(x, 0.f);
      }
      a[m][ks] = f;
    }
  }
}

template <int NET>
__device__ __forceinline__ void build_frags_V(const Lds& L, int wv, int lo16, int grp,
                                              v16h a[2][2]) {
#pragma unroll
  for (int m = 0; m < 2; ++m) {
    int r = m * 16 + lo16;
    float Vr = L.svV[wv][r];
#pragma unroll
    for (int ks = 0; ks < 2; ++ks) {
      v16h f;
#pragma unroll
      for (int e = 0; e < 16; ++e) {
        int n = ks * 32 + (e >> 3) * 16 + grp * 8 + (e & 7);
        float x = Vr * L.wi[NET][n] + L.bi[NET][n];
        f[e] = (_Float16)fmaxf(x, 0.f);
      }
      a[m][ks] = f;
    }
  }
}

// ---------------- small VALU nets ----------------
__device__ __forceinline__ void cvv_eval(const Lds& L, float tp, float S, float* out26) {
  float h1[30], h2[30];
#pragma unroll
  for (int j = 0; j < 30; ++j)
    h1[j] = fmaxf(tp * L.cvv_wi[j] + S * L.cvv_wi[30 + j] + L.cvv_bi[j], 0.f);
#pragma unroll
  for (int j = 0; j < 30; ++j) {
    float s = L.cvv_bh[0][j];
#pragma unroll
    for (int k = 0; k < 30; ++k) s += h1[k] * L.cvv_wh[0][k * 30 + j];
    h2[j] = fmaxf(s, 0.f);
  }
#pragma unroll
  for (int j = 0; j < 30; ++j) {
    float s = L.cvv_bh[1][j];
#pragma unroll
    for (int k = 0; k < 30; ++k) s += h2[k] * L.cvv_wh[1][k * 30 + j];
    h1[j] = fmaxf(s, 0.f);
  }
#pragma unroll
  for (int k = 0; k < 26; ++k) {
    float s = L.cvv_bo[k];
#pragma unroll
    for (int j = 0; j < 30; ++j) s += h1[j] * L.cvv_wo[j * 26 + k];
    out26[k] = s;
  }
}

__device__ __forceinline__ float cve_eval(const Lds& L, float tp, float V, const float* P) {
  float g1[20], g2[20];
#pragma unroll
  for (int j = 0; j < 20; ++j)
    g1[j] = fmaxf(P[j] + tp * L.cve_wi[j] + V * L.cve_wi[66 * 20 + j] + L.cve_bi[j], 0.f);
#pragma unroll
  for (int j = 0; j < 20; ++j) {
    float s = L.cve_bh[0][j];
#pragma unroll
    for (int k = 0; k < 20; ++k) s += g1[k] * L.cve_wh[0][k * 20 + j];
    g2[j] = fmaxf(s, 0.f);
  }
#pragma unroll
  for (int j = 0; j < 20; ++j) {
    float s = L.cve_bh[1][j];
#pragma unroll
    for (int k = 0; k < 20; ++k) s += g2[k] * L.cve_wh[1][k * 20 + j];
    g1[j] = fmaxf(s, 0.f);
  }
  float s = L.cve_bo[0];
#pragma unroll
  for (int j = 0; j < 20; ++j) s += g1[j] * L.cve_wo[j];
  return s;
}

#if HAVE_TDM
// Issue one TDM bulk copy: n contiguous fp32 elements global -> LDS (1-D tile).
// 6-arg builtin form: (g0, g1, g2, g3, g4, cpol) with unused groups zero/NULL.
__device__ __forceinline__ void tdm_load_1d(const float* src, unsigned lds_byte_off, int n) {
  unsigned long long ga = (unsigned long long)(uintptr_t)src;
  u32x4 g0;
  g0[0] = 1u;                                   // count=1 (valid), user mode, no gather
  g0[1] = lds_byte_off;                         // lds_addr (bytes)
  g0[2] = (unsigned)(ga & 0xffffffffu);         // global_addr[31:0]
  g0[3] = (unsigned)((ga >> 32) & 0x01ffffffu)  // global_addr[56:32]
        | (2u << 30);                           // type = 2 ("image")
  i32x8 g1;
  g1[0] = 0x20000;                              // workgroup_mask=0, data_size=2 (4B)
  g1[1] = (int)(((unsigned)n & 0xffffu) << 16); // tensor_dim0[15:0] at bits 63:48
  g1[2] = (int)((1u << 16) | ((unsigned)n >> 16)); // tensor_dim0[31:16], tensor_dim1=1
  g1[3] = (int)(((unsigned)n & 0xffffu) << 16); // tile_dim0 = n (bits 127:112)
  g1[4] = 1;                                    // tile_dim1 = 1, tile_dim2 = 0
  g1[5] = n;                                    // tensor_dim0_stride[31:0]
  g1[6] = 0;
  g1[7] = 0;                                    // tensor_dim1_stride = 0
  i32x4 gz4 = {0, 0, 0, 0};
  i32x8 gz8 = {0, 0, 0, 0, 0, 0, 0, 0};
  __builtin_amdgcn_tensor_load_to_lds(g0, g1, gz4, gz4, gz8, 0);
}
#endif

// ---------------- main persistent MC kernel ----------------
__global__ __launch_bounds__(TPB, 1)
void lsv_mc_kernel(KArgs A) {
  extern __shared__ char smem_raw[];
  Lds& L = *reinterpret_cast<Lds*>(smem_raw);
  const int tid = threadIdx.x, lane = tid & 31, wv = tid >> 5;
  const int lo16 = lane & 15, grp = lane >> 4;
  const int row = blockIdx.x * TPB + tid;

  if (tid < 65) L.tg[tid] = A.tg[tid];

  const float* whg[3] = {A.dWh, A.vWh, A.wWh};
  const float* wig[3] = {A.dWi, A.vWi, A.wWi};
  const float* big_[3] = {A.dbi, A.vbi, A.wbi};
  const float* bhg[3] = {A.dbh, A.vbh, A.wbh};
  const float* wog[3] = {A.dWo, A.vWo, A.wWo};
  const float* bog[3] = {A.dbo, A.vbo, A.wbo};
  const int wiN[3] = {192, 64, 64};

  float S = A.S0[0];
  float rho_t = tanhf(A.rho[0]);
  float s2 = sqrtf(fmaxf(1.f - rho_t * rho_t, 0.f));
  float V = 0.5f / (1.f + expf(-A.v0[0]));
  float run_max = S, cv_ex = 0.f;
  float accv[26];
#pragma unroll
  for (int k = 0; k < 26; ++k) accv[k] = 0.f;
  float P[20];
#pragma unroll
  for (int j = 0; j < 20; ++j) P[j] = 0.f;

  L.path[tid][0] = S;
  A.out[OFF_PATH + row * 65] = S;
  A.out[OFF_VAR + row * 65] = V;

  int cur_idx = -1;
  for (int i = 1; i <= NSTEPS; ++i) {
    const int idx = (i - 1) >> 5;
    if (idx != cur_idx) {  // period switch: (re)load weights, rebuild cve accumulator
      cur_idx = idx;
#pragma unroll
      for (int net = 0; net < 3; ++net) {
        __syncthreads();  // previous consumers / staging users done
#if HAVE_TDM
        // TDM bulk DMA: 3 hidden layers of this net (12288 fp32) -> LDS staging
        if (wv == 0) {
          tdm_load_1d(whg[net] + idx * 3 * 4096,
                      (unsigned)(uintptr_t)&L.stage[0], 3 * 4096);
          __builtin_amdgcn_s_wait_tensorcnt((short)0);
        }
        __syncthreads();  // staging visible to all waves
#endif
        for (int layer = 0; layer < 3; ++layer) {
#if HAVE_TDM
          const float* src = &L.stage[layer * 4096];
#else
          const float* src = whg[net] + (idx * 3 + layer) * 4096;
#endif
          for (int t = tid; t < 4096; t += TPB) {
            int e = t & 15, ln = (t >> 4) & 31, nt = (t >> 9) & 3, ks = t >> 11;
            int n = nt * 16 + (ln & 15);
            int k = ks * 32 + (e >> 3) * 16 + (ln >> 4) * 8 + (e & 7);
            L.wh[net][layer][t] = (_Float16)src[k * 64 + n];
          }
        }
        cpyf(L.wi[net], wig[net] + idx * wiN[net], wiN[net], tid);
        cpyf(L.bi[net], big_[net] + idx * 64, 64, tid);
        cpyf(&L.bh[net][0][0], bhg[net] + idx * 192, 192, tid);
        cpyf(L.wo[net], wog[net] + idx * 64, 64, tid);
        if (tid == 0) L.bo[net] = bog[net][idx];
      }
      cpyf(L.cvv_wi, A.cWi + idx * 60, 60, tid);
      cpyf(L.cvv_bi, A.cbi + idx * 30, 30, tid);
      cpyf(&L.cvv_wh[0][0], A.cWh + idx * 1800, 1800, tid);
      cpyf(&L.cvv_bh[0][0], A.cbh + idx * 60, 60, tid);
      cpyf(L.cvv_wo, A.cWo + idx * 780, 780, tid);
      cpyf(L.cvv_bo, A.cbo + idx * 26, 26, tid);
      cpyf(L.cve_wi, A.eWi + idx * 1340, 1340, tid);
      cpyf(L.cve_bi, A.ebi + idx * 20, 20, tid);
      cpyf(&L.cve_wh[0][0], A.eWh + idx * 800, 800, tid);
      cpyf(&L.cve_bh[0][0], A.ebh + idx * 40, 40, tid);
      cpyf(L.cve_wo, A.eWo + idx * 20, 20, tid);
      if (tid == 0) L.cve_bo[0] = A.ebo[idx];
      __syncthreads();
#pragma unroll
      for (int j = 0; j < 20; ++j) P[j] = 0.f;
      for (int c = 0; c < i; ++c) {  // replay path columns already set
        float pc = L.path[tid][c];
#pragma unroll
        for (int j = 0; j < 20; ++j) P[j] += pc * L.cve_wi[(1 + c) * 20 + j];
      }
    }

    const float tp = L.tg[i - 1], tc = L.tg[i];
    const float hdt = tc - tp, sqh = sqrtf(hdt);
    if (i < NSTEPS) __builtin_prefetch(&A.z[row * NSTEPS + i], 0, 1);
    const float zw = A.z[row * NSTEPS + (i - 1)];
    const float zv = A.z2[row * NSTEPS + (i - 1)];
    const float dW = sqh * zw;
    const float dB = rho_t * dW + s2 * sqh * zv;

    // publish per-row state for WMMA operand builders (wave-local)
    L.svS[wv][lane] = S;
    L.svV[wv][lane] = V;
    __builtin_amdgcn_wave_barrier();

    v16h a[2][2];
    build_frags_diff(L, wv, lo16, grp, tp, a);
    const float diff = softplusf(mlp64<0>(L, wv, lane, a));
    build_frags_V<1>(L, wv, lo16, grp, a);
    const float drv = mlp64<1>(L, wv, lane, a);
    build_frags_V<2>(L, wv, lo16, grp, a);
    const float dfv = softplusf(mlp64<2>(L, wv, lane, a));

    float co[26];
    cvv_eval(L, tp, S, co);
    const float ce = cve_eval(L, tp, V, P);

    const float disc = expf(-RATE_C * tp);
    const float cf = disc * S * diff * dW;
#pragma unroll
    for (int k = 0; k < 26; ++k) accv[k] += cf * co[k];
    cv_ex += cf * ce;

    const float S_new = S + RATE_C * S * hdt / (1.f + RATE_C * S * sqh)
                          + S * diff * dW / (1.f + S * diff * sqh);
    const float V_new = V + drv * hdt + dfv * dB;
    S = S_new;
    V = fmaxf(V_new, 0.f);
    L.path[tid][i] = S;
    A.out[OFF_PATH + row * 65 + i] = S;
    A.out[OFF_VAR + row * 65 + i] = V;
#pragma unroll
    for (int j = 0; j < 20; ++j) P[j] += S * L.cve_wi[(1 + i) * 20 + j];
    run_max = fmaxf(run_max, S);

    if (i == 32 || i == 64) {  // vanilla pricing at maturity
      const int m = (i == 32) ? 0 : 1;
      const float dtc = expf(-RATE_C * tc);
      float pr[13];
#pragma unroll
      for (int k = 0; k < 13; ++k) {
        float cv = (m == 0) ? accv[k] : accv[13 + k];
        pr[k] = dtc * fmaxf(S - (0.7f + 0.05f * (float)k), 0.f) - cv;
      }
      __syncthreads();
      if (tid < 64) L.red[tid] = 0.f;
      __syncthreads();
#pragma unroll
      for (int k = 0; k < 13; ++k) {
        atomicAdd(&L.red[k], pr[k]);
        atomicAdd(&L.red[26 + k], pr[k] * pr[k]);
      }
      __syncthreads();
      if (tid < 13) {
        atomicAdd(&A.ws[m * 13 + tid], L.red[tid]);
        atomicAdd(&A.ws[26 + m * 13 + tid], L.red[26 + tid]);
      }
    }
  }

  // exotic payoff + reductions
  const float exo = run_max - S;
  const float discT = expf(-RATE_C * L.tg[NSTEPS]);
  const float a_ = discT * exo;
  const float ep = a_ - cv_ex;
  A.out[OFF_EXO + row] = ep;
  __syncthreads();
  if (tid < 64) L.red[tid] = 0.f;
  __syncthreads();
  atomicAdd(&L.red[0], a_);
  atomicAdd(&L.red[1], ep);
  atomicAdd(&L.red[2], ep * ep);
  __syncthreads();
  if (tid == 0) {
    atomicAdd(&A.ws[52], L.red[0]);
    atomicAdd(&A.ws[53], L.red[1]);
    atomicAdd(&A.ws[54], L.red[2]);
  }
}

__global__ void finalize_kernel(float* out, float* ws) {
  const int t = threadIdx.x;
  const float N = (float)MC_N;
  if (t < 26) {
    float s = ws[t], ss = ws[26 + t];
    out[OFF_PMAT + t] = s / N;
    out[OFF_VPM + t] = (ss - s * s / N) / (N - 1.f);
  }
  if (t == 0) {
    float sp = ws[53], ssp = ws[54];
    out[OFF_MEAN] = sp / N;
    out[OFF_VARP] = (ssp - sp * sp / N) / (N - 1.f);
    ws[55] = ws[52] / N;  // mean(discT * exo) for error kernel
  }
}

__global__ void error_kernel(float* out, const float* ws) {
  const int r = blockIdx.x * TPB + threadIdx.x;
  out[OFF_ERR + r] = out[OFF_EXO + r] - ws[55];
}

extern "C" void kernel_launch(void* const* d_in, const int* in_sizes, int n_in,
                              void* d_out, int out_size, void* d_ws, size_t ws_size,
                              hipStream_t stream) {
  (void)in_sizes; (void)n_in; (void)out_size; (void)ws_size;
  KArgs A;
  A.S0 = (const float*)d_in[0];  A.z   = (const float*)d_in[1];
  A.z2 = (const float*)d_in[2];  A.tg  = (const float*)d_in[3];
  A.v0 = (const float*)d_in[4];  A.rho = (const float*)d_in[5];
  // pytree flattening of each param dict is key-sorted: Wh, Wi, Wo, bh, bi, bo
  A.dWh=(const float*)d_in[6];  A.dWi=(const float*)d_in[7];  A.dWo=(const float*)d_in[8];
  A.dbh=(const float*)d_in[9];  A.dbi=(const float*)d_in[10]; A.dbo=(const float*)d_in[11];
  A.vWh=(const float*)d_in[12]; A.vWi=(const float*)d_in[13]; A.vWo=(const float*)d_in[14];
  A.vbh=(const float*)d_in[15]; A.vbi=(const float*)d_in[16]; A.vbo=(const float*)d_in[17];
  A.wWh=(const float*)d_in[18]; A.wWi=(const float*)d_in[19]; A.wWo=(const float*)d_in[20];
  A.wbh=(const float*)d_in[21]; A.wbi=(const float*)d_in[22]; A.wbo=(const float*)d_in[23];
  A.cWh=(const float*)d_in[24]; A.cWi=(const float*)d_in[25]; A.cWo=(const float*)d_in[26];
  A.cbh=(const float*)d_in[27]; A.cbi=(const float*)d_in[28]; A.cbo=(const float*)d_in[29];
  A.eWh=(const float*)d_in[30]; A.eWi=(const float*)d_in[31]; A.eWo=(const float*)d_in[32];
  A.ebh=(const float*)d_in[33]; A.ebi=(const float*)d_in[34]; A.ebo=(const float*)d_in[35];
  A.out = (float*)d_out;
  A.ws  = (float*)d_ws;

  (void)hipMemsetAsync(d_ws, 0, 64 * sizeof(float), stream);
  lsv_mc_kernel<<<MC_N / TPB, TPB, sizeof(Lds), stream>>>(A);
  finalize_kernel<<<1, 32, 0, stream>>>((float*)d_out, (float*)d_ws);
  error_kernel<<<MC_N / TPB, TPB, 0, stream>>>((float*)d_out, (const float*)d_ws);
}
